// Interaction_58145267253392
// MI455X (gfx1250) — compile-verified, matrix operand order checked by reading, not run on hardware
//
#include <hip/hip_runtime.h>
#include <math.h>

#define N_NODES 10000
#define N_EDGES 160000
#define HID 128

typedef __attribute__((ext_vector_type(2)))  float    v2f;
typedef __attribute__((ext_vector_type(8)))  float    v8f;
typedef __attribute__((ext_vector_type(16))) _Float16 v16h;
typedef __attribute__((ext_vector_type(4)))  unsigned v4u;
typedef __attribute__((ext_vector_type(8)))  int      v8i;
typedef __attribute__((ext_vector_type(4)))  int      v4i;

// ---------------------------------------------------------------------------
// Kernel 0: convert edge-MLP weights to f16 once per launch (tiny).
// ---------------------------------------------------------------------------
__global__ void k_wconv(const float* __restrict__ Ws0, const float* __restrict__ Ws1,
                        const float* __restrict__ Ws2, _Float16* __restrict__ wh) {
  int j = blockIdx.x * blockDim.x + threadIdx.x;
  if (j < 4096)            wh[j] = (_Float16)Ws0[j];
  else if (j < 36864)      wh[j] = (_Float16)Ws1[j - 4096];
  else if (j < 135168)     wh[j] = (_Float16)Ws2[j - 36864];
}

// ---------------------------------------------------------------------------
// Kernel 1: per-node pre-pass. TDM-stage X tile (8 nodes) to LDS, clin with Wp
// via f32 WMMA, add bias, frobenius-normalize, decompose into structured parts
// i[N,128], a[N,128,3], s[N,128,6].
// ---------------------------------------------------------------------------
__global__ void __launch_bounds__(256) k_node_pre(
    const float* __restrict__ Xg, const float* __restrict__ Wp, const float* __restrict__ bp,
    float* __restrict__ gI, float* __restrict__ gA, float* __restrict__ gS) {
  __shared__ float lds[8 * HID * 9];   // 36 KB: X tile, then reused as T[128][72]
  const int n0   = blockIdx.x * 8;
  const int tid  = threadIdx.x;
  const int lane = tid & 31, wv = tid >> 5;
  const int m = lane & 15, h = lane >> 4, nc = lane & 15;

  // ---- stage X[n0..n0+8) (contiguous 9216 f32) into LDS via Tensor Data Mover
#if defined(__AMDGCN__) && __has_builtin(__builtin_amdgcn_tensor_load_to_lds)
  if (wv == 0) {
    unsigned long long ga = (unsigned long long)(size_t)(Xg + (size_t)n0 * (HID * 9));
    const unsigned elems = 8u * HID * 9;            // 9216 4-byte elements
    v4u g0; v8i g1; v4i gz; v8i gz8;
    g0[0] = 1u;                                     // D# count = 1 (valid)
    g0[1] = 0u;                                     // lds_addr: offset 0
    g0[2] = (unsigned)(ga & 0xffffffffu);           // global_addr lo
    g0[3] = (unsigned)((ga >> 32) & 0x01ffffffu) | (2u << 30);  // addr hi + type=2
    g1[0] = (int)(2u << 16);                        // data_size = 2 -> 4B elements
    g1[1] = (int)((elems & 0xffffu) << 16);         // tensor_dim0[15:0]
    g1[2] = (int)(((elems >> 16) & 0xffffu) | (1u << 16)); // dim0 hi, tensor_dim1=1
    g1[3] = (int)((elems & 0xffffu) << 16);         // tile_dim0 = 9216
    g1[4] = 0;                                      // tile_dim1/2 unused
    g1[5] = (int)elems;                             // tensor_dim0_stride
    g1[6] = 0; g1[7] = 0;
    gz[0] = gz[1] = gz[2] = gz[3] = 0;
    gz8[0] = gz8[1] = gz8[2] = gz8[3] = gz8[4] = gz8[5] = gz8[6] = gz8[7] = 0;
    __builtin_amdgcn_tensor_load_to_lds(g0, g1, gz, gz, gz8, 0);
    __builtin_amdgcn_s_wait_tensorcnt(0);
  }
  __syncthreads();
#else
  for (int idx = tid; idx < 8 * HID * 9; idx += 256)
    lds[idx] = Xg[(size_t)n0 * (HID * 9) + idx];
  __syncthreads();
#endif

  // ---- A fragments (this wave's 16 rows of Wp), hoisted across N-tiles
  v2f afr[32];
  const float* Wrow = Wp + (wv * 16 + m) * HID;
#pragma unroll
  for (int ks = 0; ks < 32; ++ks) {
    afr[ks].x = Wrow[ks * 4 + 2 * h];
    afr[ks].y = Wrow[ks * 4 + 2 * h + 1];
  }

  // ---- 5 N-tiles of 16 cover 72 columns (8 nodes x 9 tensor slots)
  v8f acc[5];
#pragma unroll
  for (int nt = 0; nt < 5; ++nt) {
    int col = nt * 16 + nc;
    bool cv = col < 72;
    int node = cv ? (col / 9) : 0;
    int ij   = cv ? (col % 9) : 0;
    const float* Bb = lds + node * (HID * 9) + ij;
    v8f z = {}; acc[nt] = z;
#pragma unroll
    for (int ks = 0; ks < 32; ++ks) {
      int c0 = ks * 4 + 2 * h;
      v2f b;
      b.x = cv ? Bb[c0 * 9] : 0.0f;
      b.y = cv ? Bb[(c0 + 1) * 9] : 0.0f;
      acc[nt] = __builtin_amdgcn_wmma_f32_16x16x4_f32(false, afr[ks], false, b,
                                                      (short)0, acc[nt], false, false);
    }
  }
  __syncthreads();    // all X reads done -> reuse LDS as T[o][col]
#pragma unroll
  for (int nt = 0; nt < 5; ++nt) {
    int col = nt * 16 + nc;
    if (col < 72) {
#pragma unroll
      for (int r = 0; r < 8; ++r) {
        int o = wv * 16 + r + 8 * h;
        lds[o * 72 + col] = acc[nt][r] + bp[o];
      }
    }
  }
  __syncthreads();

  // ---- normalize + decompose: 8 nodes x 128 channels
  for (int it = 0; it < 4; ++it) {
    int idx = tid + it * 256;
    int n = idx >> 7, o = idx & 127;
    float t[9]; float n2 = 0.0f;
#pragma unroll
    for (int ij = 0; ij < 9; ++ij) { t[ij] = lds[o * 72 + n * 9 + ij]; n2 += t[ij] * t[ij]; }
    float sc = 1.0f / (fmaxf(n2, 0.01f) + 1.0f);
#pragma unroll
    for (int ij = 0; ij < 9; ++ij) t[ij] *= sc;
    float pi3 = (t[0] + t[4] + t[8]) * (1.0f / 3.0f);
    int g = (n0 + n) * 128 + o;
    gI[g] = pi3;
    gA[g * 3 + 0] = 0.5f * (t[1] - t[3]);
    gA[g * 3 + 1] = 0.5f * (t[2] - t[6]);
    gA[g * 3 + 2] = 0.5f * (t[5] - t[7]);
    gS[g * 6 + 0] = t[0] - pi3;
    gS[g * 6 + 1] = 0.5f * (t[1] + t[3]);
    gS[g * 6 + 2] = 0.5f * (t[2] + t[6]);
    gS[g * 6 + 3] = t[4] - pi3;
    gS[g * 6 + 4] = 0.5f * (t[5] + t[7]);
    gS[g * 6 + 5] = t[8] - pi3;
  }
}

// ---------------------------------------------------------------------------
// Kernel 2: structured clin GEMM: O[n,o,k] = sum_c W[o,c] * B[n,c,k] (f32 WMMA).
// KP = 1 (diag), 3 (antisym), 6 (sym). One block = one 16-col N-tile, 8 waves
// cover the 8 M-tiles (128 outputs).
// ---------------------------------------------------------------------------
__global__ void __launch_bounds__(256) k_clin(const float* __restrict__ W,
                                              const float* __restrict__ Bp,
                                              float* __restrict__ Op,
                                              int KP, int ncols) {
  int lane = threadIdx.x & 31, wv = threadIdx.x >> 5;
  int ntn  = (ncols + 15) >> 4;
  int tile = blockIdx.x * 8 + wv;
  if (tile >= ntn * 8) return;
  int mt = tile & 7, nt = tile >> 3;
  int m = lane & 15, h = lane >> 4;
  int col = nt * 16 + (lane & 15);
  bool cv = col < ncols;
  int node = cv ? col / KP : 0;
  int kk   = cv ? col - node * KP : 0;
  int cb   = (node * 128) * KP + kk;
  const float* Wrow = W + (mt * 16 + m) * 128;
  v8f acc = {};
#pragma unroll 4
  for (int ks = 0; ks < 32; ++ks) {
    int c0 = ks * 4 + 2 * h;
    v2f a, b;
    a.x = Wrow[c0]; a.y = Wrow[c0 + 1];
    b.x = cv ? Bp[cb + c0 * KP] : 0.0f;
    b.y = cv ? Bp[cb + (c0 + 1) * KP] : 0.0f;
    acc = __builtin_amdgcn_wmma_f32_16x16x4_f32(false, a, false, b, (short)0, acc, false, false);
  }
  if (cv) {
#pragma unroll
    for (int r = 0; r < 8; ++r) {
      int o = mt * 16 + r + 8 * h;
      Op[(node * 128 + o) * KP + kk] = acc[r];
    }
  }
}

// ---------------------------------------------------------------------------
// Kernel 3: fused edge pipeline. 16 edges/block: 3-layer MLP via f16 WMMA
// (32->128->256->384) with SiLU + cosine cutoff, then structured gather of the
// transformed parts yI/yA/yS at dst and coalesced atomic scatter into src.
// ---------------------------------------------------------------------------
__global__ void __launch_bounds__(256) k_edge(
    const float* __restrict__ eattr, const float* __restrict__ ew, const int* __restrict__ eidx,
    const _Float16* __restrict__ Ws0h, const float* __restrict__ bs0,
    const _Float16* __restrict__ Ws1h, const float* __restrict__ bs1,
    const _Float16* __restrict__ Ws2h, const float* __restrict__ bs2,
    const float* __restrict__ yI, const float* __restrict__ yA, const float* __restrict__ yS,
    float* __restrict__ mI, float* __restrict__ mA, float* __restrict__ mS) {
  __shared__ _Float16 sAttr[16 * 32];
  __shared__ _Float16 sH0[16 * 128];
  __shared__ _Float16 sH1[16 * 256];
  __shared__ float    sEA[16 * 384];
  __shared__ float    sC[16];
  __shared__ int      sSrc[16], sDst[16];
  const int e0 = blockIdx.x * 16;
  const int tid = threadIdx.x;
  const int lane = tid & 31, wv = tid >> 5;
  const int m = lane & 15, h = lane >> 4, ec = lane & 15;

  for (int idx = tid; idx < 16 * 32; idx += 256)
    sAttr[idx] = (_Float16)eattr[e0 * 32 + idx];
  if (tid < 16) {
    float wgt = ew[e0 + tid];
    sC[tid] = 0.5f * (cosf(wgt * 3.14159265358979323846f / 5.0f) + 1.0f)
            * (wgt < 5.0f ? 1.0f : 0.0f);
    sSrc[tid] = eidx[e0 + tid];
    sDst[tid] = eidx[N_EDGES + e0 + tid];
  }
  __syncthreads();

  // ---- L0: [128,32] x [32,16], K=32 in one WMMA
  {
    const _Float16* Ar = Ws0h + (wv * 16 + m) * 32;
    const _Float16* Br = sAttr + ec * 32;
    v16h a, b; v8f acc = {};
#pragma unroll
    for (int j = 0; j < 8; ++j) {
      a[j] = Ar[8 * h + j];   a[8 + j] = Ar[16 + 8 * h + j];
      b[j] = Br[8 * h + j];   b[8 + j] = Br[16 + 8 * h + j];
    }
    acc = __builtin_amdgcn_wmma_f32_16x16x32_f16(false, a, false, b, (short)0, acc, false, false);
#pragma unroll
    for (int r = 0; r < 8; ++r) {
      int o = wv * 16 + r + 8 * h;
      float v = acc[r] + bs0[o];
      sH0[ec * 128 + o] = (_Float16)(v / (1.0f + expf(-v)));
    }
  }
  __syncthreads();

  // ---- L1: [256,128] x [128,16], 2 M-tiles per wave, 4 K-steps
#pragma unroll
  for (int t = 0; t < 2; ++t) {
    int mt = wv + 8 * t;
    const _Float16* Ar = Ws1h + (mt * 16 + m) * 128;
    const _Float16* Br = sH0 + ec * 128;
    v8f acc = {};
#pragma unroll
    for (int ks = 0; ks < 4; ++ks) {
      int k0 = ks * 32;
      v16h a, b;
#pragma unroll
      for (int j = 0; j < 8; ++j) {
        a[j] = Ar[k0 + 8 * h + j];   a[8 + j] = Ar[k0 + 16 + 8 * h + j];
        b[j] = Br[k0 + 8 * h + j];   b[8 + j] = Br[k0 + 16 + 8 * h + j];
      }
      acc = __builtin_amdgcn_wmma_f32_16x16x32_f16(false, a, false, b, (short)0, acc, false, false);
    }
#pragma unroll
    for (int r = 0; r < 8; ++r) {
      int o = mt * 16 + r + 8 * h;
      float v = acc[r] + bs1[o];
      sH1[ec * 256 + o] = (_Float16)(v / (1.0f + expf(-v)));
    }
  }
  __syncthreads();

  // ---- L2: [384,256] x [256,16], 3 M-tiles per wave, 8 K-steps -> ea * C
#pragma unroll
  for (int t = 0; t < 3; ++t) {
    int mt = wv * 3 + t;
    const _Float16* Ar = Ws2h + (mt * 16 + m) * 256;
    const _Float16* Br = sH1 + ec * 256;
    v8f acc = {};
#pragma unroll
    for (int ks = 0; ks < 8; ++ks) {
      int k0 = ks * 32;
      v16h a, b;
#pragma unroll
      for (int j = 0; j < 8; ++j) {
        a[j] = Ar[k0 + 8 * h + j];   a[8 + j] = Ar[k0 + 16 + 8 * h + j];
        b[j] = Br[k0 + 8 * h + j];   b[8 + j] = Br[k0 + 16 + 8 * h + j];
      }
      acc = __builtin_amdgcn_wmma_f32_16x16x32_f16(false, a, false, b, (short)0, acc, false, false);
    }
#pragma unroll
    for (int r = 0; r < 8; ++r) {
      int o = mt * 16 + r + 8 * h;
      float v = acc[r] + bs2[o];
      sEA[ec * 384 + o] = (v / (1.0f + expf(-v))) * sC[ec];
    }
  }
  __syncthreads();

  // ---- structured gather (dst) + coalesced atomic scatter (src): 10 f/channel
  for (int e = 0; e < 16; ++e) {
    const int src = sSrc[e], dst = sDst[e];
    const float* ea = sEA + e * 384;          // ea[c*3 + r]
    for (int j = tid; j < 1280; j += 256) {
      if (j < 128) {
        int c = j;
        atomicAdd(&mI[src * 128 + c], ea[c * 3 + 0] * yI[dst * 128 + c]);
      } else if (j < 512) {
        int off = j - 128; int c = off / 3;
        atomicAdd(&mA[src * 384 + off], ea[c * 3 + 1] * yA[dst * 384 + off]);
      } else {
        int off = j - 512; int c = off / 6;
        atomicAdd(&mS[src * 768 + off], ea[c * 3 + 2] * yS[dst * 768 + off]);
      }
    }
  }
}

// ---------------------------------------------------------------------------
// Kernel 4: per (node,channel): P = M@Y + Y@M, decompose, renorm -> overwrite
// Y-part buffers (safe: each thread reads its Y before writing its slots).
// ---------------------------------------------------------------------------
__device__ __forceinline__ void recon3(float* T, float pi, const float* a, const float* s) {
  T[0] =  pi + s[0];   T[1] =  a[0] + s[1]; T[2] =  a[1] + s[2];
  T[3] = -a[0] + s[1]; T[4] =  pi + s[3];   T[5] =  a[2] + s[4];
  T[6] = -a[1] + s[2]; T[7] = -a[2] + s[4]; T[8] =  pi + s[5];
}

__global__ void k_product(const float* __restrict__ mI, const float* __restrict__ mA,
                          const float* __restrict__ mS,
                          float* __restrict__ yI, float* __restrict__ yA, float* __restrict__ yS) {
  int id = blockIdx.x * blockDim.x + threadIdx.x;
  if (id >= N_NODES * 128) return;
  float M[9], Y[9], P[9];
  recon3(M, mI[id], &mA[id * 3], &mS[id * 6]);
  recon3(Y, yI[id], &yA[id * 3], &yS[id * 6]);
#pragma unroll
  for (int i = 0; i < 3; ++i)
#pragma unroll
    for (int j = 0; j < 3; ++j) {
      float v = 0.0f;
#pragma unroll
      for (int k = 0; k < 3; ++k)
        v += M[i * 3 + k] * Y[k * 3 + j] + Y[i * 3 + k] * M[k * 3 + j];
      P[i * 3 + j] = v;
    }
  float n2 = 0.0f;
#pragma unroll
  for (int ij = 0; ij < 9; ++ij) n2 += P[ij] * P[ij];
  float sc = 1.0f / (fmaxf(n2, 0.01f) + 1.0f);
  float pi3 = (P[0] + P[4] + P[8]) * (1.0f / 3.0f);
  yI[id] = pi3 * sc;
  yA[id * 3 + 0] = 0.5f * (P[1] - P[3]) * sc;
  yA[id * 3 + 1] = 0.5f * (P[2] - P[6]) * sc;
  yA[id * 3 + 2] = 0.5f * (P[5] - P[7]) * sc;
  yS[id * 6 + 0] = (P[0] - pi3) * sc;
  yS[id * 6 + 1] = 0.5f * (P[1] + P[3]) * sc;
  yS[id * 6 + 2] = 0.5f * (P[2] + P[6]) * sc;
  yS[id * 6 + 3] = (P[4] - pi3) * sc;
  yS[id * 6 + 4] = 0.5f * (P[5] + P[7]) * sc;
  yS[id * 6 + 5] = (P[8] - pi3) * sc;
}

// ---------------------------------------------------------------------------
// Kernel 5: out = reconstruct(Xn parts) + reconstruct(dX parts)
// ---------------------------------------------------------------------------
__global__ void k_assemble(const float* __restrict__ gI, const float* __restrict__ gA,
                           const float* __restrict__ gS,
                           const float* __restrict__ dI, const float* __restrict__ dA,
                           const float* __restrict__ dS, float* __restrict__ out) {
  int id = blockIdx.x * blockDim.x + threadIdx.x;
  if (id >= N_NODES * 128) return;
  float T[9], D[9];
  recon3(T, gI[id], &gA[id * 3], &gS[id * 6]);
  recon3(D, dI[id], &dA[id * 3], &dS[id * 6]);
#pragma unroll
  for (int ij = 0; ij < 9; ++ij) out[id * 9 + ij] = T[ij] + D[ij];
}

// ---------------------------------------------------------------------------
extern "C" void kernel_launch(void* const* d_in, const int* in_sizes, int n_in,
                              void* d_out, int out_size, void* d_ws, size_t ws_size,
                              hipStream_t stream) {
  const float* X     = (const float*)d_in[0];
  const float* ew    = (const float*)d_in[1];
  const float* eattr = (const float*)d_in[2];
  const float* Wp    = (const float*)d_in[3];
  const float* bp    = (const float*)d_in[4];
  const float* Ws0   = (const float*)d_in[5];
  const float* bs0   = (const float*)d_in[6];
  const float* Ws1   = (const float*)d_in[7];
  const float* bs1   = (const float*)d_in[8];
  const float* Ws2   = (const float*)d_in[9];
  const float* bs2   = (const float*)d_in[10];
  const int*   eidx  = (const int*)d_in[11];
  const float* Wt0   = (const float*)d_in[12];
  const float* Wt1   = (const float*)d_in[13];
  const float* Wt2   = (const float*)d_in[14];
  const float* Wt3   = (const float*)d_in[15];
  const float* Wt4   = (const float*)d_in[16];
  const float* Wt5   = (const float*)d_in[17];
  float* out = (float*)d_out;

  // workspace layout (floats); region3 (msg) is reused for dX parts
  float* ws = (float*)d_ws;
  const size_t nf = (size_t)N_NODES * 128;
  float *gI = ws,          *gA = ws + nf,      *gS = ws + 4 * nf;   // Xn parts
  float *yI = ws + 10 * nf, *yA = ws + 11 * nf, *yS = ws + 14 * nf; // Y parts -> P parts
  float *mI = ws + 20 * nf, *mA = ws + 21 * nf, *mS = ws + 24 * nf; // msg -> dX parts
  _Float16* wh = (_Float16*)(ws + 30 * nf);
  _Float16 *Ws0h = wh, *Ws1h = wh + 4096, *Ws2h = wh + 36864;

  k_wconv   <<<dim3(528), dim3(256), 0, stream>>>(Ws0, Ws1, Ws2, wh);
  k_node_pre<<<dim3(N_NODES / 8), dim3(256), 0, stream>>>(X, Wp, bp, gI, gA, gS);

  // Y = clin(I,Wt0)+clin(A,Wt1)+clin(S,Wt2), stored as structured parts
  k_clin<<<dim3((N_NODES * 1 + 15) / 16), dim3(256), 0, stream>>>(Wt0, gI, yI, 1, N_NODES * 1);
  k_clin<<<dim3((N_NODES * 3 + 15) / 16), dim3(256), 0, stream>>>(Wt1, gA, yA, 3, N_NODES * 3);
  k_clin<<<dim3((N_NODES * 6 + 15) / 16), dim3(256), 0, stream>>>(Wt2, gS, yS, 6, N_NODES * 6);

  (void)hipMemsetAsync(mI, 0, 10 * nf * sizeof(float), stream);
  k_edge<<<dim3(N_EDGES / 16), dim3(256), 0, stream>>>(eattr, ew, eidx, Ws0h, bs0, Ws1h, bs1,
                                                       Ws2h, bs2, yI, yA, yS, mI, mA, mS);

  k_product<<<dim3((unsigned)((nf + 255) / 256)), dim3(256), 0, stream>>>(mI, mA, mS, yI, yA, yS);

  // dX parts (into the msg region, no longer needed)
  k_clin<<<dim3((N_NODES * 1 + 15) / 16), dim3(256), 0, stream>>>(Wt3, yI, mI, 1, N_NODES * 1);
  k_clin<<<dim3((N_NODES * 3 + 15) / 16), dim3(256), 0, stream>>>(Wt4, yA, mA, 3, N_NODES * 3);
  k_clin<<<dim3((N_NODES * 6 + 15) / 16), dim3(256), 0, stream>>>(Wt5, yS, mS, 6, N_NODES * 6);

  k_assemble<<<dim3((unsigned)((nf + 255) / 256)), dim3(256), 0, stream>>>(gI, gA, gS, mI, mA, mS, out);
}
